// SampledSofmax_20220706029753
// MI455X (gfx1250) — matrix operand order repaired and checked by reading, not compiled, not run on hardware
//
#include <hip/hip_runtime.h>
#include <hip/hip_bf16.h>
#include <math.h>

// Problem constants (from reference): B=2, T=1024, D=1024, U=50257, S=8192
#define BT    2048          // B*T rows
#define DIM   1024          // feature dim (K of GEMM)
#define UVOC  50257         // vocab
#define SNEG  8192          // sampled negatives
#define LOG_S 9.010913347279289f   // ln(8192)

typedef __attribute__((ext_vector_type(16))) __bf16 v16bf;
typedef __attribute__((ext_vector_type(8)))  float  v8f;
typedef unsigned int u32x4 __attribute__((ext_vector_type(4)));
typedef int          i32x8 __attribute__((ext_vector_type(8)));
typedef int          i32x4 __attribute__((ext_vector_type(4)));

#if __has_builtin(__builtin_amdgcn_tensor_load_to_lds) && \
    __has_builtin(__builtin_amdgcn_s_wait_tensorcnt)
#define USE_TDM 1
#else
#define USE_TDM 0
#endif

// ---------------- helpers ----------------
__device__ __forceinline__ unsigned short f2bf_rne(float f) {
  unsigned u = __float_as_uint(f);
  unsigned r = 0x7FFFu + ((u >> 16) & 1u);        // round-to-nearest-even
  return (unsigned short)((u + r) >> 16);
}
__device__ __forceinline__ unsigned pack2bf(float lo, float hi) {
  return (unsigned)f2bf_rne(lo) | ((unsigned)f2bf_rne(hi) << 16);
}
// log of log-uniform sampling probability (tf.random.log_uniform_candidate_sampler)
__device__ __forceinline__ float logq_logu(int id) {
  float idf = (float)id;
  return logf(log1pf(1.0f / (idf + 1.0f)) / logf((float)(UVOC + 1)));
}

// ---------------- conversion / gather kernels ----------------
__global__ void cvt_f32_to_bf16_kernel(const float* __restrict__ src,
                                       unsigned* __restrict__ dst, int npairs) {
  int i = blockIdx.x * blockDim.x + threadIdx.x;
  int stride = gridDim.x * blockDim.x;
  for (; i < npairs; i += stride) {
    float2 f = ((const float2*)src)[i];
    dst[i] = pack2bf(f.x, f.y);
  }
}

__global__ void gather_rows_bf16_kernel(const float* __restrict__ W,
                                        const int* __restrict__ sampled,
                                        unsigned* __restrict__ dst /* S x D/2 */) {
  int i = blockIdx.x * blockDim.x + threadIdx.x;
  int stride = gridDim.x * blockDim.x;
  const int rowdw = DIM / 2;
  for (; i < SNEG * rowdw; i += stride) {
    int row = i / rowdw;
    int c   = i - row * rowdw;
    int src = sampled[row];
    float2 f = ((const float2*)(W + (size_t)src * DIM))[c];
    dst[i] = pack2bf(f.x, f.y);
  }
}

__global__ void sampled_bias_kernel(const float* __restrict__ bias,
                                    const int* __restrict__ sampled,
                                    float* __restrict__ bs) {
  int s = blockIdx.x * blockDim.x + threadIdx.x;
  if (s < SNEG) {
    int id = sampled[s];
    bs[s] = bias[id] - (LOG_S + logq_logu(id));
  }
}

// ---------------- true logits (per-row gather dot) ----------------
__global__ __launch_bounds__(256)
void true_logits_kernel(const float* __restrict__ x, const float* __restrict__ W,
                        const float* __restrict__ bias, const int* __restrict__ targets,
                        float* __restrict__ tl) {
  const int n = blockIdx.x;
  const int t = targets[n];
  const float* xr = x + (size_t)n * DIM;
  const float* wr = W + (size_t)t * DIM;
  float s = 0.f;
  for (int d = threadIdx.x; d < DIM; d += 256) s += xr[d] * wr[d];
  __shared__ float red[256];
  red[threadIdx.x] = s; __syncthreads();
  for (int o = 128; o > 0; o >>= 1) {
    if (threadIdx.x < o) red[threadIdx.x] += red[threadIdx.x + o];
    __syncthreads();
  }
  if (threadIdx.x == 0)
    tl[n] = red[0] + bias[t] - (LOG_S + logq_logu(t));
}

// ---------------- TDM 2D tile stage (D# per CDNA5 ISA) ----
#define LDS_ROW_DW 20                 // 64B bf16 data + 16B pad per tile row
#define TILE_DW    (128 * LDS_ROW_DW) // dwords per staged tile buffer

#if USE_TDM
__device__ __forceinline__ void tdm_stage_2d(unsigned ldsAddr,
                                             const unsigned* __restrict__ gsrc,
                                             unsigned rowsRemain) {
  // bf16 tile: tile_dim0 = 32 elems (64B), tile_dim1 = 128 rows,
  // tensor row stride = DIM elems; LDS pad: 4 dwords after every 16 dwords.
  unsigned long long ga = (unsigned long long)(uintptr_t)gsrc;
  u32x4 g0;
  g0.x = 1u;                                   // count=1, user descriptor
  g0.y = ldsAddr;                              // lds_addr (bytes)
  g0.z = (unsigned)ga;                         // global_addr[31:0]
  g0.w = (unsigned)(ga >> 32) | (2u << 30);    // global_addr[56:32] | type=2
  i32x8 g1;
  g1[0] = (1 << 16)      // data_size = 2 bytes
        | (1 << 20)      // pad_enable (LDS dest padding)
        | (3 << 22)      // pad_interval: 16 dwords
        | (3 << 25);     // pad_amount: 4 dwords (16B)
  g1[1] = (int)(((unsigned)DIM & 0xFFFFu) << 16);              // tensor_dim0[15:0]
  g1[2] = (int)(((unsigned)DIM >> 16) | ((rowsRemain & 0xFFFFu) << 16));
  g1[3] = (int)((rowsRemain >> 16) | (32u << 16));             // tile_dim0 = 32
  g1[4] = 128;                                                 // tile_dim1=128, tile_dim2=0
  g1[5] = DIM;                                                 // tensor_dim0_stride lo
  g1[6] = 0;
  g1[7] = 0;
  i32x4 z4 = {0, 0, 0, 0};                     // groups 2/3 unused (2D tensor)
  i32x8 z8 = {0, 0, 0, 0, 0, 0, 0, 0};         // extra group (6-arg toolchain form)
  __builtin_amdgcn_tensor_load_to_lds(g0, g1, z4, z4, z8, 0);
}
#endif

// ---------------- bf16 WMMA GEMM: out[n,u] = sum_k A[n,k]*Bm[u,k] (+bias) ----------
// 128x128 tile per block, BK=32 (one v_wmma_f32_16x16x32_bf16 k-slab), 8 waves,
// double-buffered LDS with issue-ahead TDM staging (TENSORcnt-tracked).
// mode 0: write logits + bias[u] to outLogits[n*Ucols+u]
// mode 1: fused sampled-softmax epilogue: mask accidental hits, exp(), row-sum
//         over the 128-col tile -> outPartial[n*gridDim.y + blockIdx.y]
__global__ __launch_bounds__(256)
void wmma_gemm_kernel(const unsigned* __restrict__ Abf,      // [2048 x D/2] packed bf16
                      const unsigned* __restrict__ Bbf,      // [Ucols x D/2] packed bf16
                      const float* __restrict__ biasAdd,     // [Ucols]
                      float* __restrict__ outLogits,
                      float* __restrict__ outPartial,
                      const int* __restrict__ targets,
                      const int* __restrict__ sampled,
                      int Ucols, int mode) {
  __shared__ __align__(16) unsigned lds[4 * TILE_DW];   // A0 A1 B0 B1

  const int tid     = threadIdx.x;
  const int wave    = tid >> 5;        // 0..7, wave32
  const int lane    = tid & 31;
  const int l16     = lane & 15;
  const int halfsel = lane >> 4;       // 0: lanes 0-15, 1: lanes 16-31
  const int rowBase = blockIdx.x * 128;
  const int colBase = blockIdx.y * 128;
  const int rowdw   = DIM / 2;

  v8f acc[8];
  #pragma unroll
  for (int g = 0; g < 8; ++g)
    #pragma unroll
    for (int j = 0; j < 8; ++j) acc[g][j] = 0.0f;

  // buffer bases computed with runtime arithmetic (no constant generic-pointer
  // arrays: those become addrspacecast static initializers that ld.lld rejects)
  auto stage = [&](int kb, int buf) {
    unsigned* aDst = lds + buf * TILE_DW;
    unsigned* bDst = lds + (2 + buf) * TILE_DW;
#if USE_TDM
    if (wave == 0) {     // TDM ignores EXEC; issue once, wave-uniform guard
      tdm_stage_2d((unsigned)(uintptr_t)aDst,
                   Abf + (size_t)rowBase * rowdw + (kb >> 1),
                   (unsigned)(BT - rowBase));
      tdm_stage_2d((unsigned)(uintptr_t)bDst,
                   Bbf + (size_t)colBase * rowdw + (kb >> 1),
                   (unsigned)(Ucols - colBase));
    }
#else
    const int kdw = kb >> 1;
    for (int q = tid; q < 512; q += 256) {
      const int r = q >> 2;            // tile row 0..127
      const int c = q & 3;             // 16B quad within 64B row
      uint4 av = *(const uint4*)(Abf + (size_t)(rowBase + r) * rowdw + kdw + c * 4);
      *(uint4*)(aDst + r * LDS_ROW_DW + c * 4) = av;
      uint4 bv = make_uint4(0u, 0u, 0u, 0u);
      if (colBase + r < Ucols) {
        bv = *(const uint4*)(Bbf + (size_t)(colBase + r) * rowdw + kdw + c * 4);
        if (kb + 32 < DIM)
          __builtin_prefetch(Bbf + (size_t)(colBase + r) * rowdw + kdw + 16 + c * 4, 0, 3);
      }
      *(uint4*)(bDst + r * LDS_ROW_DW + c * 4) = bv;
    }
#endif
  };

  union Frag { uint4 u[2]; v16bf v; };

  stage(0, 0);
  int cur = 0;
  for (int kb = 0; kb < DIM; kb += 32, cur ^= 1) {
#if USE_TDM
    if (wave == 0) __builtin_amdgcn_s_wait_tensorcnt(0);   // slab kb landed
#endif
    __syncthreads();                                       // visible to all waves
    if (kb + 32 < DIM) stage(kb + 32, cur ^ 1);            // issue-ahead next slab

    const unsigned* aCur = lds + cur * TILE_DW;
    const unsigned* bCur = lds + (2 + cur) * TILE_DW;

    // ---- A fragment (16-bit A 16x32 layout, ISA 7.12.2) ----
    Frag af;
    {
      const unsigned* arow = aCur + (wave * 16 + l16) * LDS_ROW_DW;
      af.u[0] = *(const uint4*)(arow + (halfsel ? 4 : 0));    // K 0-7 / 8-15
      af.u[1] = *(const uint4*)(arow + (halfsel ? 12 : 8));   // K 16-23 / 24-31
    }
    // ---- all 8 B fragments up front (distinct regs -> no per-WMMA dscnt wait) ----
    Frag bf[8];
    #pragma unroll
    for (int g = 0; g < 8; ++g) {
      const unsigned* brow = bCur + (g * 16 + l16) * LDS_ROW_DW;
      bf[g].u[0] = *(const uint4*)(brow + (halfsel ? 4 : 0));
      bf[g].u[1] = *(const uint4*)(brow + (halfsel ? 12 : 8));
    }
    #pragma unroll
    for (int g = 0; g < 8; ++g)
      acc[g] = __builtin_amdgcn_wmma_f32_16x16x32_bf16(
          false, af.v, false, bf[g].v, (short)0, acc[g], false, false);

    __syncthreads();     // readers done before buf[cur] is re-staged at kb+64
  }

  if (mode == 0) {
    // epilogue: logits + bias
    #pragma unroll
    for (int g = 0; g < 8; ++g) {
      const int col = colBase + g * 16 + l16;
      if (col < Ucols) {
        const float b = biasAdd[col];
        #pragma unroll
        for (int v = 0; v < 8; ++v) {
          const int row = rowBase + wave * 16 + halfsel * 8 + v;   // C/D layout
          outLogits[(size_t)row * Ucols + col] = acc[g][v] + b;
        }
      }
    }
  } else {
    // fused sampled-softmax epilogue: exp(logit + b_s), accidental-hit mask,
    // deterministic per-tile row sums (no float atomics).
    int tv[8];
    #pragma unroll
    for (int v = 0; v < 8; ++v)
      tv[v] = targets[rowBase + wave * 16 + halfsel * 8 + v];
    float rs[8];
    #pragma unroll
    for (int v = 0; v < 8; ++v) rs[v] = 0.f;
    #pragma unroll
    for (int g = 0; g < 8; ++g) {
      const int col = colBase + g * 16 + l16;
      const float b = biasAdd[col];
      const int sc  = sampled[col];
      #pragma unroll
      for (int v = 0; v < 8; ++v) {
        const float lg = acc[g][v] + b;
        rs[v] += (tv[v] == sc) ? 0.f : __expf(lg);
      }
    }
    #pragma unroll
    for (int v = 0; v < 8; ++v) {
      float s = rs[v];
      s += __shfl_xor(s, 1);
      s += __shfl_xor(s, 2);
      s += __shfl_xor(s, 4);
      s += __shfl_xor(s, 8);           // lanes 0..15 / 16..31 each hold one row
      if (l16 == 0) {
        const int row = rowBase + wave * 16 + halfsel * 8 + v;
        outPartial[(size_t)row * gridDim.y + blockIdx.y] = s;
      }
    }
  }
}

// ---------------- in-place row softmax over U ----------------
__global__ __launch_bounds__(512)
void softmax_rows_kernel(float* __restrict__ logits) {
  const int n = blockIdx.x;
  float* p = logits + (size_t)n * UVOC;
  __shared__ float red[512];
  float m = -3.4e38f;
  for (int i = threadIdx.x; i < UVOC; i += 512) m = fmaxf(m, p[i]);
  red[threadIdx.x] = m; __syncthreads();
  for (int o = 256; o > 0; o >>= 1) {
    if (threadIdx.x < o) red[threadIdx.x] = fmaxf(red[threadIdx.x], red[threadIdx.x + o]);
    __syncthreads();
  }
  m = red[0]; __syncthreads();
  float s = 0.f;
  for (int i = threadIdx.x; i < UVOC; i += 512) s += __expf(p[i] - m);
  red[threadIdx.x] = s; __syncthreads();
  for (int o = 256; o > 0; o >>= 1) {
    if (threadIdx.x < o) red[threadIdx.x] += red[threadIdx.x + o];
    __syncthreads();
  }
  const float inv = 1.0f / red[0];
  for (int i = threadIdx.x; i < UVOC; i += 512) p[i] = __expf(p[i] - m) * inv;
}

// ---------------- loss finalize ----------------
__global__ __launch_bounds__(256)
void loss_finalize_kernel(const float* __restrict__ partial, int ntiles,
                          const float* __restrict__ tl, float* __restrict__ outLoss) {
  float acc = 0.f;
  for (int n = threadIdx.x; n < BT; n += 256) {
    float es = 0.f;
    const float* pr = partial + (size_t)n * ntiles;
    for (int c = 0; c < ntiles; ++c) es += pr[c];
    const float t = tl[n];
    es += __expf(t);                    // true-class slot
    acc += __logf(es) - t;              // logsumexp - true_logit (max-free: logits bounded)
  }
  __shared__ float red[256];
  red[threadIdx.x] = acc; __syncthreads();
  for (int o = 128; o > 0; o >>= 1) {
    if (threadIdx.x < o) red[threadIdx.x] += red[threadIdx.x + o];
    __syncthreads();
  }
  if (threadIdx.x == 0) *outLoss = red[0] / (float)BT;
}

// ---------------- host launcher ----------------
extern "C" void kernel_launch(void* const* d_in, const int* in_sizes, int n_in,
                              void* d_out, int out_size, void* d_ws, size_t ws_size,
                              hipStream_t stream) {
  const float* x       = (const float*)d_in[0];   // [B,T,D]
  const float* W       = (const float*)d_in[1];   // [U,D]
  const float* bias    = (const float*)d_in[2];   // [U]
  const int*   targets = (const int*)d_in[3];     // [B,T]
  const int*   sampled = (const int*)d_in[4];     // [S]

  float* probs   = (float*)d_out;                       // [BT, U]
  float* lossOut = probs + (size_t)BT * UVOC;           // scalar

  // workspace carve-out (256B aligned)
  char* ws = (char*)d_ws;
  size_t off = 0;
  auto carve = [&](size_t bytes) -> void* {
    void* p = ws + off;
    off = (off + bytes + 255) & ~(size_t)255;
    return p;
  };
  unsigned* Xb      = (unsigned*)carve((size_t)BT   * DIM / 2 * 4);   //   4 MB bf16 X
  unsigned* Wb      = (unsigned*)carve((size_t)UVOC * DIM / 2 * 4);   // 103 MB bf16 W (L2-resident)
  unsigned* Wsb     = (unsigned*)carve((size_t)SNEG * DIM / 2 * 4);   //  16 MB bf16 W[sampled]
  float*    bsAdj   = (float*)carve((size_t)SNEG * 4);
  float*    tl      = (float*)carve((size_t)BT * 4);
  float*    partial = (float*)carve((size_t)BT * (SNEG / 128) * 4);

  // 1) precision conversion (one pass; bf16 W fits in 192MB L2 for GEMM re-reads)
  {
    int n2 = BT * DIM / 2;
    cvt_f32_to_bf16_kernel<<<(n2 + 255) / 256, 256, 0, stream>>>(x, Xb, n2);
  }
  {
    int n2 = UVOC * DIM / 2;
    cvt_f32_to_bf16_kernel<<<(n2 + 255) / 256, 256, 0, stream>>>(W, Wb, n2);
  }
  gather_rows_bf16_kernel<<<(SNEG * DIM / 2 + 255) / 256, 256, 0, stream>>>(W, sampled, Wsb);
  sampled_bias_kernel<<<(SNEG + 255) / 256, 256, 0, stream>>>(bias, sampled, bsAdj);

  // 2) full-vocab logits GEMM (WMMA bf16, f32 accum) -> d_out, then softmax in-place
  {
    dim3 grid(BT / 128, (UVOC + 127) / 128);   // 16 x 393
    wmma_gemm_kernel<<<grid, 256, 0, stream>>>(Xb, Wb, bias, probs, nullptr,
                                               nullptr, nullptr, UVOC, 0);
  }
  softmax_rows_kernel<<<BT, 512, 0, stream>>>(probs);

  // 3) sampled-softmax loss
  true_logits_kernel<<<BT, 256, 0, stream>>>(x, W, bias, targets, tl);
  {
    dim3 grid(BT / 128, SNEG / 128);           // 16 x 64
    wmma_gemm_kernel<<<grid, 256, 0, stream>>>(Xb, Wsb, bsAdj, nullptr, partial,
                                               targets, sampled, SNEG, 1);
  }
  loss_finalize_kernel<<<1, 256, 0, stream>>>(partial, SNEG / 128, tl, lossOut);
}
